// SelfAttention_2405181686173
// MI455X (gfx1250) — matrix operand order repaired and checked by reading, not compiled, not run on hardware
//
#include <hip/hip_runtime.h>
#include <stdint.h>

// ---------------- problem constants (from reference) ----------------
constexpr int Bb = 4;
constexpr int Cc = 256;
constexpr int Nn = 64 * 64;   // 4096
constexpr int Dd = 32;        // C/8
constexpr int NT = Nn / 16;   // 256 16-wide tiles along N
constexpr int JB = Nn / 32;   // 128 32-key blocks
constexpr int CT = Cc / 16;   // 16 channel tiles
constexpr int KC = Cc / 32;   // 8 K-chunks of 32 channels

typedef __attribute__((ext_vector_type(16))) _Float16 v16h;
typedef __attribute__((ext_vector_type(8)))  float    v8f;

union V16H { uint4 u[2]; v16h h; };

static __device__ __forceinline__ uint16_t f16b(float a) {
    union { _Float16 h; uint16_t u; } c; c.h = (_Float16)a; return c.u;
}
static __device__ __forceinline__ uint32_t pack2h(float a, float b) {
    return (uint32_t)f16b(a) | ((uint32_t)f16b(b) << 16);
}

// A-operand image coordinates for a 16x32 f16 tile, element (m, k):
//   lanes 0-15 hold K 0-7 (g0-3) & 16-23 (g4-7); lanes 16-31 hold K 8-15 & 24-31
static __device__ __forceinline__ void a_coord(int m, int k, int& ln, int& g) {
    ln = m + 16 * (((k & 15) >= 8) ? 1 : 0);
    g  = (k < 16) ? ((k & 7) >> 1) : (4 + ((k & 7) >> 1));
}

// =====================================================================
// pack_x: x (f32, B x C x N) -> f16 B-operand images per 32-channel chunk
//   xB[b][kc][nt][lane][g]  (element (kk, nn): lane = nn+16*((kk&15)>=8),
//                            g = (kk<16?0:4)+((kk&7)>>1), pos = kk&1)
// =====================================================================
__global__ void __launch_bounds__(256) pack_x_kernel(
    const float* __restrict__ x, uint32_t* __restrict__ xB)
{
    const int n  = blockIdx.x * 256 + threadIdx.x;
    const int cc = blockIdx.y;
    const int b  = blockIdx.z;
    const float v = x[((size_t)b * Cc + cc) * Nn + n];

    const int kc = cc >> 5, kk = cc & 31;
    const int nt = n >> 4,  nn = n & 15;
    const int ln  = nn + 16 * (((kk & 15) >= 8) ? 1 : 0);
    const int g   = (kk < 16) ? ((kk & 7) >> 1) : (4 + ((kk & 7) >> 1));
    const int pos = kk & 1;
    uint16_t* dst = (uint16_t*)(xB + (((size_t)(b * KC + kc) * NT + nt) * 32 + ln) * 8 + g);
    dst[pos] = f16b(v);
}

// =====================================================================
// pack_w: W (rows x 256 f32) -> f16 A-operand images
//   dst[rt][kc][lane][g], rt = output-row tile, kc = 32-wide K chunk
//   grid.x = rows, block = 256 (one thread per element)
// =====================================================================
__global__ void __launch_bounds__(256) pack_w_kernel(
    const float* __restrict__ W, uint32_t* __restrict__ dst)
{
    const int r = blockIdx.x;
    const int c = threadIdx.x;
    const int rt = r >> 4, m = r & 15;
    const int kc = c >> 5, kk = c & 31;
    int ln, g;
    a_coord(m, kk, ln, g);
    uint16_t* d = (uint16_t*)(dst + ((size_t)(rt * KC + kc) * 32 + ln) * 8 + g);
    d[kk & 1] = f16b(W[(size_t)r * Cc + c]);
}

// =====================================================================
// proj_wmma: q/k/v projections on the matrix units.
//   block = 288 threads = 9 waves; blockIdx.x = nt*2 + half.
//   wave ot = half*9 + w:  ot==0 -> q (both d-subtiles, transposed A-image)
//                          ot==1 -> k (both d-subtiles, B-image, in-register)
//                          ot>=2 -> v channel tile ct = ot-2 (B-image of v^T)
// =====================================================================
__global__ void __launch_bounds__(288) proj_wmma_kernel(
    const uint32_t* __restrict__ xB,
    const uint32_t* __restrict__ WqA, const float* __restrict__ bq,
    const uint32_t* __restrict__ WkA, const float* __restrict__ bk,
    const uint32_t* __restrict__ WvA, const float* __restrict__ bv,
    uint32_t* __restrict__ qA, uint32_t* __restrict__ kB,
    uint32_t* __restrict__ vB)
{
    __shared__ uint16_t qstage[16 * 32];   // only wave ot==0 uses this

    const int lane = threadIdx.x & 31;
    const int w    = threadIdx.x >> 5;
    const int nt   = blockIdx.x >> 1;
    const int half = blockIdx.x & 1;
    const int b    = blockIdx.y;
    const int ot   = half * 9 + w;         // 0..17

    const int mrow = (lane >= 16) ? 8 : 0;
    const v8f zacc = {};

    // per-kc B-operand images of x for this nt
    const uint32_t* xBt = xB + ((size_t)b * KC * NT + nt) * 256 + lane * 8;

    if (ot >= 2) {
        // ---------------- v: one 16-channel tile ----------------
        const int ct = ot - 2;
        v8f acc = zacc;
#pragma unroll
        for (int kc = 0; kc < KC; ++kc) {
            V16H A, Bx;
            const uint4* ap = (const uint4*)(WvA + ((size_t)(ct * KC + kc)) * 256 + lane * 8);
            A.u[0] = ap[0]; A.u[1] = ap[1];
            const uint4* bp = (const uint4*)(xBt + (size_t)kc * NT * 256);
            Bx.u[0] = bp[0]; Bx.u[1] = bp[1];
            acc = __builtin_amdgcn_wmma_f32_16x16x32_f16(
                false, A.h, false, Bx.h, (short)0, acc, false, false);
        }
        // pack to B-operand image of v^T (K = pixels, N = channels)
        const int jb  = nt >> 1;
        const int jhi = nt & 1;
        uint32_t* vT = vB + ((size_t)((b * JB + jb) * CT + ct)) * 256;
#pragma unroll
        for (int r = 0; r < 8; ++r) {
            const float val = acc[r] + bv[ct * 16 + r + mrow];
            const float nb  = __shfl_xor(val, 1, 32);
            if ((lane & 1) == 0) {
                const int lane_t = r + mrow + 16 * jhi;
                const int g      = (lane & 15) >> 1;
                vT[lane_t * 8 + g] = pack2h(val, nb);
            }
        }
        return;
    }

    // ---------------- q or k: both 16-row d-subtiles ----------------
    const uint32_t* WA = (ot == 0) ? WqA : WkA;
    const float*    bs = (ot == 0) ? bq  : bk;
    v8f acc0 = zacc, acc1 = zacc;
#pragma unroll
    for (int kc = 0; kc < KC; ++kc) {
        V16H A0, A1, Bx;
        const uint4* a0 = (const uint4*)(WA + ((size_t)(0 * KC + kc)) * 256 + lane * 8);
        const uint4* a1 = (const uint4*)(WA + ((size_t)(1 * KC + kc)) * 256 + lane * 8);
        A0.u[0] = a0[0]; A0.u[1] = a0[1];
        A1.u[0] = a1[0]; A1.u[1] = a1[1];
        const uint4* bp = (const uint4*)(xBt + (size_t)kc * NT * 256);
        Bx.u[0] = bp[0]; Bx.u[1] = bp[1];
        acc0 = __builtin_amdgcn_wmma_f32_16x16x32_f16(
            false, A0.h, false, Bx.h, (short)0, acc0, false, false);
        acc1 = __builtin_amdgcn_wmma_f32_16x16x32_f16(
            false, A1.h, false, Bx.h, (short)0, acc1, false, false);
    }

    if (ot == 1) {
        // k -> B-operand image (K = d, N = keys), packed in-register
        uint32_t* kT = kB + ((size_t)(b * NT + nt)) * 256;
#pragma unroll
        for (int st = 0; st < 2; ++st) {
            const int lane_t = (lane & 15) + 2 * mrow;
#pragma unroll
            for (int r = 0; r < 8; r += 2) {
                const float v0 = ((st == 0) ? acc0[r]     : acc1[r])     + bs[st * 16 + r + mrow];
                const float v1 = ((st == 0) ? acc0[r + 1] : acc1[r + 1]) + bs[st * 16 + r + 1 + mrow];
                const int g = (r >> 1) + 4 * st;
                kT[lane_t * 8 + g] = pack2h(v0, v1);
            }
        }
    } else {
        // q -> transpose (d x n -> n x d) through LDS, emit A-operand image
#pragma unroll
        for (int st = 0; st < 2; ++st) {
#pragma unroll
            for (int r = 0; r < 8; ++r) {
                const float v = ((st == 0) ? acc0[r] : acc1[r]) + bs[st * 16 + r + mrow];
                qstage[(lane & 15) * 32 + st * 16 + r + mrow] = f16b(v);
            }
        }
        const uint8_t* Pb = (const uint8_t*)qstage;
        const int m   = lane & 15;
        const int off = (lane < 16) ? 0 : 16;
        const uint4 a0 = *(const uint4*)(Pb + m * 64 + off);
        const uint4 a1 = *(const uint4*)(Pb + m * 64 + 32 + off);
        uint4* qT = (uint4*)(qA + ((size_t)(b * NT + nt)) * 256 + lane * 8);
        qT[0] = a0; qT[1] = a1;
    }
}

// =====================================================================
// attn: fused flash-style attention
//   one wave = 16 queries x 128 channels (2 waves cover all 256 channels)
// =====================================================================
__global__ void __launch_bounds__(256) attn_kernel(
    const float* __restrict__ x, const float* __restrict__ gamma,
    const uint32_t* __restrict__ qA, const uint32_t* __restrict__ kB,
    const uint32_t* __restrict__ vB, float* __restrict__ out)
{
    __shared__ float smem[8][2048];  // 64 KB: per-wave 8KB, P-stage then O-stage

    const int lane = threadIdx.x & 31;
    const int w    = threadIdx.x >> 5;
    const int gw   = blockIdx.x * 8 + w;
    const int qi    = gw >> 1;           // query tile 0..255
    const int chalf = gw & 1;            // channel half (128 ch each)
    const int b     = blockIdx.y;

    V16H Aq;
    {
        const uint4* p = (const uint4*)(qA + ((size_t)(b * NT + qi)) * 256 + lane * 8);
        Aq.u[0] = p[0]; Aq.u[1] = p[1];
    }

    const uint32_t* kTb = kB + (size_t)b * NT * 256;
    const uint32_t* vTb = vB + (size_t)b * JB * CT * 256;
    const v8f zacc = {};

    // -------- pass 1: row maxima over all keys --------
    float m8[8];
#pragma unroll
    for (int r = 0; r < 8; ++r) m8[r] = -3.0e38f;

    for (int kj = 0; kj < NT; ++kj) {
        V16H Bk;
        const uint4* p = (const uint4*)(kTb + (size_t)kj * 256 + lane * 8);
        Bk.u[0] = p[0]; Bk.u[1] = p[1];
        v8f S = __builtin_amdgcn_wmma_f32_16x16x32_f16(
            false, Aq.h, false, Bk.h, (short)0, zacc, false, false);
#pragma unroll
        for (int r = 0; r < 8; ++r) m8[r] = fmaxf(m8[r], S[r]);
    }
#pragma unroll
    for (int r = 0; r < 8; ++r) {
        m8[r] = fmaxf(m8[r], __shfl_xor(m8[r], 1, 32));
        m8[r] = fmaxf(m8[r], __shfl_xor(m8[r], 2, 32));
        m8[r] = fmaxf(m8[r], __shfl_xor(m8[r], 4, 32));
        m8[r] = fmaxf(m8[r], __shfl_xor(m8[r], 8, 32));
    }

    // -------- pass 2: exp, row sums, P @ V^T --------
    float l8[8];
    v8f o[8];
#pragma unroll
    for (int r = 0; r < 8; ++r) l8[r] = 0.f;
#pragma unroll
    for (int ct = 0; ct < 8; ++ct) o[ct] = zacc;

    uint16_t* P = (uint16_t*)&smem[w][0];      // 16 x 32 f16 staging
    const int mrow = (lane >= 16) ? 8 : 0;
    const int colb = lane & 15;
    const int cb0  = chalf * 8;

    for (int jb = 0; jb < JB; ++jb) {
        V16H Bk0, Bk1;
        {
            const uint4* p0 = (const uint4*)(kTb + (size_t)(2 * jb) * 256 + lane * 8);
            const uint4* p1 = (const uint4*)(kTb + (size_t)(2 * jb + 1) * 256 + lane * 8);
            Bk0.u[0] = p0[0]; Bk0.u[1] = p0[1];
            Bk1.u[0] = p1[0]; Bk1.u[1] = p1[1];
        }
        v8f S0 = __builtin_amdgcn_wmma_f32_16x16x32_f16(
            false, Aq.h, false, Bk0.h, (short)0, zacc, false, false);
        v8f S1 = __builtin_amdgcn_wmma_f32_16x16x32_f16(
            false, Aq.h, false, Bk1.h, (short)0, zacc, false, false);

#pragma unroll
        for (int r = 0; r < 8; ++r) {
            const float p0 = __expf(S0[r] - m8[r]);
            const float p1 = __expf(S1[r] - m8[r]);
            l8[r] += p0 + p1;
            P[(r + mrow) * 32 + colb]      = f16b(p0);
            P[(r + mrow) * 32 + colb + 16] = f16b(p1);
        }

        // gather P in A-operand layout: two contiguous 16B runs per lane
        V16H Ap;
        {
            const uint8_t* Pb = (const uint8_t*)P;
            const int m   = lane & 15;
            const int off = (lane < 16) ? 0 : 32;
            Ap.u[0] = *(const uint4*)(Pb + m * 64 + off);
            Ap.u[1] = *(const uint4*)(Pb + m * 64 + 16 + off);
        }

        const uint32_t* vt = vTb + (size_t)jb * CT * 256;
#pragma unroll
        for (int ct = 0; ct < 8; ++ct) {
            V16H Bv;
            const uint4* p = (const uint4*)(vt + (size_t)(cb0 + ct) * 256 + lane * 8);
            Bv.u[0] = p[0]; Bv.u[1] = p[1];
            o[ct] = __builtin_amdgcn_wmma_f32_16x16x32_f16(
                false, Ap.h, false, Bv.h, (short)0, o[ct], false, false);
        }
    }

    // -------- epilogue: normalize, transpose via LDS, residual, store --------
#pragma unroll
    for (int r = 0; r < 8; ++r) {
        l8[r] += __shfl_xor(l8[r], 1, 32);
        l8[r] += __shfl_xor(l8[r], 2, 32);
        l8[r] += __shfl_xor(l8[r], 4, 32);
        l8[r] += __shfl_xor(l8[r], 8, 32);
        l8[r] = 1.0f / l8[r];
    }

    float* OS = &smem[w][0];  // 128 channels x 16 queries (f32) = 8 KB
#pragma unroll
    for (int ct = 0; ct < 8; ++ct) {
        const int cl = ct * 16 + colb;
#pragma unroll
        for (int r = 0; r < 8; ++r)
            OS[cl * 16 + r + mrow] = o[ct][r] * l8[r];
    }

    const float gm = gamma[0];
    const int q0 = qi * 16;
#pragma unroll
    for (int it = 0; it < 4; ++it) {
        const int cl = it * 32 + lane;             // 0..127
        const int c  = chalf * 128 + cl;
        const float*  xr   = x   + (((size_t)b * Cc + c) * Nn + q0);
        float*        orow = out + (((size_t)b * Cc + c) * Nn + q0);
        const float4* ov = (const float4*)&OS[cl * 16];
        const float4* xv = (const float4*)xr;
        float4*       dv = (float4*)orow;
#pragma unroll
        for (int k = 0; k < 4; ++k) {
            float4 a  = ov[k];
            float4 xx = xv[k];
            a.x = fmaf(gm, a.x, xx.x);
            a.y = fmaf(gm, a.y, xx.y);
            a.z = fmaf(gm, a.z, xx.z);
            a.w = fmaf(gm, a.w, xx.w);
            dv[k] = a;
        }
    }
}

// =====================================================================
extern "C" void kernel_launch(void* const* d_in, const int* in_sizes, int n_in,
                              void* d_out, int out_size, void* d_ws, size_t ws_size,
                              hipStream_t stream)
{
    const float* x  = (const float*)d_in[0];
    const float* Wq = (const float*)d_in[1];
    const float* bq = (const float*)d_in[2];
    const float* Wk = (const float*)d_in[3];
    const float* bk = (const float*)d_in[4];
    const float* Wv = (const float*)d_in[5];
    const float* bv = (const float*)d_in[6];
    const float* gm = (const float*)d_in[7];
    float* out = (float*)d_out;

    // workspace layout (u32 words):
    uint32_t* xB  = (uint32_t*)d_ws;                          // 4*8*256*256   = 2097152 (8 MB)
    uint32_t* qA  = xB  + (size_t)Bb * KC * NT * 256;         // 4*256*256     = 262144  (1 MB)
    uint32_t* kB  = qA  + (size_t)Bb * NT * 256;              // 4*256*256     = 262144  (1 MB)
    uint32_t* vB  = kB  + (size_t)Bb * NT * 256;              // 4*128*16*256  = 2097152 (8 MB)
    uint32_t* WqA = vB  + (size_t)Bb * JB * CT * 256;         // 2*8*256       = 4096
    uint32_t* WkA = WqA + (size_t)2 * KC * 256;               // 4096
    uint32_t* WvA = WkA + (size_t)2 * KC * 256;               // 16*8*256      = 32768

    pack_x_kernel<<<dim3(Nn / 256, Cc, Bb), 256, 0, stream>>>(x, xB);
    pack_w_kernel<<<dim3(Dd), 256, 0, stream>>>(Wq, WqA);
    pack_w_kernel<<<dim3(Dd), 256, 0, stream>>>(Wk, WkA);
    pack_w_kernel<<<dim3(Cc), 256, 0, stream>>>(Wv, WvA);
    proj_wmma_kernel<<<dim3(NT * 2, Bb), 288, 0, stream>>>(
        xB, WqA, bq, WkA, bk, WvA, bv, qA, kB, vB);
    attn_kernel<<<dim3((NT * 2) / 8, Bb), 256, 0, stream>>>(x, gm, qA, kB, vB, out);
}
